// TwoLayerTransformer_59605556133934
// MI455X (gfx1250) — matrix operand order repaired and checked by reading, not compiled, not run on hardware
//
#include <hip/hip_runtime.h>

typedef __attribute__((ext_vector_type(16))) _Float16 v16h;
typedef __attribute__((ext_vector_type(8)))  float    v8f;

#define T_LEN 4096
#define T1    4097
#define H_N   4
#define D_N   64
#define BS_N  4
#define EPSF  1e-7f
#define NRT   257          // row tiles of 16 covering 4097 rows
#define CPAD  4128         // padded c-length for XthT (multiple of 32)

__device__ __forceinline__ int imin(int a, int b) { return a < b ? a : b; }

// ---------------------------------------------------------------------------
// Kernel 1a: build the A-fragment table.
// The 16x32 A tile of the causal Toeplitz matrix depends only on the diagonal
// offset s = rt - 2*cb (0..256) and head h.  Atab[h][s][lane][e] holds the
// f16 WMMA A fragments in contiguous k order (same as B): lane = g*16+lm is row m=lm, element e
// holds K = e + 16g, value = exp(W[idx,h]) at idx = 16s + lm - K
// (0 outside [0,T]; idx==T -> exp(0)=1 per the reference clip semantics).
// The softmax denominator (prefix sum) cancels in the downstream L2-norm.
// ---------------------------------------------------------------------------
__global__ void build_atab_kernel(const float* __restrict__ W,
                                  _Float16* __restrict__ Atab)
{
    const int total = H_N * NRT * 32 * 16;         // 526,336
    const int stride = gridDim.x * blockDim.x;
    for (int i = blockIdx.x * blockDim.x + threadIdx.x; i < total; i += stride) {
        const int e    = i & 15;
        const int lane = (i >> 4) & 31;
        const int s    = (i >> 9) % NRT;
        const int h    = i / (512 * NRT);
        const int g    = lane >> 4;
        const int lm   = lane & 15;
        const int K    = e + 16 * g;
        const int idx  = 16 * s + lm - K;
        float v = 0.f;
        if (idx >= 0 && idx <= T_LEN)
            v = (idx == T_LEN) ? 1.0f : expf(W[(size_t)idx * H_N + h]);
        Atab[i] = (_Float16)v;
    }
}

// ---------------------------------------------------------------------------
// Kernel 1b: XthT[b][d][c] = f16 transpose of Xt, c padded to 4128 with zeros
// (row c==T is the zero pad row; c>T is tail padding multiplied by A=0).
// Contiguous-in-c so a B fragment is one 32B aligned load per lane.
// ---------------------------------------------------------------------------
__global__ void build_xt_kernel(const float* __restrict__ X,
                                _Float16* __restrict__ XthT)
{
    const size_t total = (size_t)BS_N * D_N * CPAD; // 1,056,768
    const int stride = gridDim.x * blockDim.x;
    for (size_t i = blockIdx.x * blockDim.x + threadIdx.x; i < total; i += (size_t)stride) {
        const int    c   = (int)(i % CPAD);
        const size_t row = i / CPAD;
        const int    d   = (int)(row % D_N);
        const size_t b   = row / D_N;
        const float  v   = (c < T_LEN) ? X[((size_t)b * T_LEN + c) * D_N + d] : 0.f;
        XthT[i] = (_Float16)v;
    }
}

// ---------------------------------------------------------------------------
// Kernel 2: u[b,h,r,:] = sum_{c<=r} exp(W[r-c,h]) * Xt[b,c,:] via WMMA.
// Block = (b, rt); wave = head h (4 waves share the same B stream -> WGP$ hits).
// Each wave: 16 rows x 64 cols = 4 accumulators; per 32-wide K chunk:
//   1 A frag (32B load, table) + 4 B frags (32B loads) + 4 v_wmma.
// ---------------------------------------------------------------------------
__global__ __launch_bounds__(128)
void toeplitz_conv_wmma(const _Float16* __restrict__ Atab,
                        const _Float16* __restrict__ XthT,
                        float* __restrict__ U)
{
    const int blk = blockIdx.x;
    const int rt  = blk % NRT;
    const int b   = blk / NRT;
    const int h    = threadIdx.x >> 5;   // wave id = head
    const int lane = threadIdx.x & 31;
    const int g    = lane >> 4;
    const int lm   = lane & 15;

    const int r_base  = rt * 16;
    const int kmax    = imin(r_base + 15, T_LEN);
    const int nchunks = (kmax >> 5) + 1;          // uniform across the block

    // A stream: s starts at rt, decreases by 2 per chunk (stays >= 0).
    const _Float16* ap = Atab + (((size_t)h * NRT + rt) * 32 + lane) * 16;
    // B stream: lane holds column n = tile*16 + lm, K = e + 16*g.
    const _Float16* bp = XthT + ((size_t)b * D_N + lm) * CPAD + 16 * g;

    v8f acc0 = {0.f,0.f,0.f,0.f,0.f,0.f,0.f,0.f};
    v8f acc1 = acc0, acc2 = acc0, acc3 = acc0;

    for (int cb = 0; cb < nchunks; ++cb) {
        const v16h af = *(const v16h*)ap;
        const v16h b0 = *(const v16h*)(bp);
        const v16h b1 = *(const v16h*)(bp + 16 * CPAD);
        const v16h b2 = *(const v16h*)(bp + 32 * CPAD);
        const v16h b3 = *(const v16h*)(bp + 48 * CPAD);
        acc0 = __builtin_amdgcn_wmma_f32_16x16x32_f16(false, af, false, b0,
                                                      (short)0, acc0, false, false);
        acc1 = __builtin_amdgcn_wmma_f32_16x16x32_f16(false, af, false, b1,
                                                      (short)0, acc1, false, false);
        acc2 = __builtin_amdgcn_wmma_f32_16x16x32_f16(false, af, false, b2,
                                                      (short)0, acc2, false, false);
        acc3 = __builtin_amdgcn_wmma_f32_16x16x32_f16(false, af, false, b3,
                                                      (short)0, acc3, false, false);
        ap -= 2 * 32 * 16;   // s -= 2
        bp += 32;            // c0 += 32
    }

    // C/D layout: VGPR i, lane group g -> row M = i + 8g, col N = lm (+16*tile).
    float* ub = U + ((size_t)b * H_N + h) * (size_t)T1 * D_N;
#pragma unroll
    for (int i = 0; i < 8; ++i) {
        const int rr = r_base + i + 8 * g;
        if (rr <= T_LEN) {
            float* rowp = ub + (size_t)rr * D_N + lm;
            rowp[0]  = acc0[i];
            rowp[16] = acc1[i];
            rowp[32] = acc2[i];
            rowp[48] = acc3[i];
        }
    }
}

// ---------------------------------------------------------------------------
// 64-thread block sum reduction (broadcast result).
// ---------------------------------------------------------------------------
__device__ float blockSum64(float v, float* red)
{
    const int tid = threadIdx.x;
    red[tid] = v; __syncthreads();
    for (int s = 32; s > 0; s >>= 1) {
        if (tid < s) red[tid] += red[tid + s];
        __syncthreads();
    }
    const float r = red[0];
    __syncthreads();
    return r;
}

// ---------------------------------------------------------------------------
// Kernel 3: per (b,t): L2-normalize u over d, build 5-vector with f32 Xt,
// normalize over the 5 heads, dot against row T -> dots[b][t][5].
// ---------------------------------------------------------------------------
__global__ __launch_bounds__(64)
void dots_kernel(const float* __restrict__ U, const float* __restrict__ X,
                 float* __restrict__ dots)
{
    __shared__ float red[64];
    const int t = blockIdx.x;
    const int b = blockIdx.y;
    const int d = threadIdx.x;

    const float* ub = U + (size_t)b * H_N * T1 * D_N;
    float ut[H_N], uT[H_N];
#pragma unroll
    for (int h = 0; h < H_N; ++h) {
        ut[h] = ub[((size_t)h * T1 + t)     * D_N + d];
        uT[h] = ub[((size_t)h * T1 + T_LEN) * D_N + d];
    }
    float nt[H_N], nT[H_N];
    for (int h = 0; h < H_N; ++h) nt[h] = sqrtf(blockSum64(ut[h] * ut[h], red)) + EPSF;
    for (int h = 0; h < H_N; ++h) nT[h] = sqrtf(blockSum64(uT[h] * uT[h], red)) + EPSF;

    const float xt = X[((size_t)b * T_LEN + t) * D_N + d];
    float wt[5], wT[5];
#pragma unroll
    for (int h = 0; h < H_N; ++h) { wt[h] = ut[h] / nt[h]; wT[h] = uT[h] / nT[h]; }
    wt[4] = xt;
    wT[4] = 0.f;                       // Xt row T is the zero pad
    float st = 0.f, sT = 0.f;
#pragma unroll
    for (int k = 0; k < 5; ++k) { st += wt[k] * wt[k]; sT += wT[k] * wT[k]; }
    st = sqrtf(st) + EPSF;
    sT = sqrtf(sT) + EPSF;
#pragma unroll
    for (int k = 0; k < 5; ++k) {
        const float dsum = blockSum64((wt[k] / st) * (wT[k] / sT), red);
        if (d == 0) dots[((size_t)b * T_LEN + t) * 5 + k] = dsum;
    }
}

// ---------------------------------------------------------------------------
// Kernel 4: p = prod over batch; 31-subset polynomial; softmax over T.
// ---------------------------------------------------------------------------
__global__ __launch_bounds__(256)
void attn_kernel(const float* __restrict__ dots, const float* __restrict__ C,
                 const float* __restrict__ a, float* __restrict__ attn)
{
    __shared__ float red[256];
    __shared__ float sc[32];
    __shared__ float bc;
    const int tid = threadIdx.x;
    if (tid < 32) sc[tid] = (tid > 0) ? C[tid] * C[tid] : 0.f;
    __syncthreads();
    const float av = a[0];

    float lg[16];
    float lmax = -3.4e38f;
    for (int j = 0; j < 16; ++j) {
        const int t = tid * 16 + j;
        float p[5];
#pragma unroll
        for (int k = 0; k < 5; ++k) {
            float pr = 1.f;
#pragma unroll
            for (int b = 0; b < BS_N; ++b)
                pr *= dots[((size_t)b * T_LEN + t) * 5 + k];
            p[k] = pr;
        }
        float kern = 0.f;
        for (int s = 1; s < 32; ++s) {
            float term = 1.f;
#pragma unroll
            for (int k = 0; k < 5; ++k)
                if ((s >> (4 - k)) & 1) term *= p[k];   // bit (4-k) selects slot k
            kern += sc[s] * term;
        }
        lg[j] = av * kern;
        lmax  = fmaxf(lmax, lg[j]);
    }
    red[tid] = lmax; __syncthreads();
    for (int s = 128; s > 0; s >>= 1) {
        if (tid < s) red[tid] = fmaxf(red[tid], red[tid + s]);
        __syncthreads();
    }
    if (tid == 0) bc = red[0];
    __syncthreads();
    const float gmax = bc;

    float lsum = 0.f;
    for (int j = 0; j < 16; ++j) lsum += expf(lg[j] - gmax);
    __syncthreads();
    red[tid] = lsum; __syncthreads();
    for (int s = 128; s > 0; s >>= 1) {
        if (tid < s) red[tid] += red[tid + s];
        __syncthreads();
    }
    if (tid == 0) bc = red[0];
    __syncthreads();
    const float gsum = bc;

    for (int j = 0; j < 16; ++j)
        attn[tid * 16 + j] = expf(lg[j] - gmax) / gsum;
}

// ---------------------------------------------------------------------------
// Kernel 5: y[b,d] = sum_t attn[t] * X[b,t,d]  (256 outputs, one block).
// ---------------------------------------------------------------------------
__global__ __launch_bounds__(256)
void out_kernel(const float* __restrict__ attn, const float* __restrict__ X,
                float* __restrict__ out)
{
    const int i = threadIdx.x;
    const int b = i >> 6, d = i & 63;
    float acc = 0.f;
    for (int t = 0; t < T_LEN; ++t)
        acc += attn[t] * X[((size_t)b * T_LEN + t) * D_N + d];
    out[i] = acc;
}

// ---------------------------------------------------------------------------
// Workspace layout (bytes, 256-aligned):
//   Atab @ 0          : 1,052,672 (f16 [4][257][32][16])
//   XthT @ 1,052,672  : 2,113,536 (f16 [4][64][4128])
//   U    @ 3,166,208  : 16,781,312 (f32 [4][4][4097][64])
//   dots @ 19,947,520 :   327,680 (f32 [4][4096][5])
//   attn @ 20,275,200 :    16,384 (f32 [4096])
// Total ~20.3 MB.
// ---------------------------------------------------------------------------
extern "C" void kernel_launch(void* const* d_in, const int* in_sizes, int n_in,
                              void* d_out, int out_size, void* d_ws, size_t ws_size,
                              hipStream_t stream)
{
    const float* X = (const float*)d_in[0];
    const float* W = (const float*)d_in[1];
    const float* a = (const float*)d_in[2];
    const float* C = (const float*)d_in[3];
    float* out = (float*)d_out;

    char* ws = (char*)d_ws;
    _Float16* Atab = (_Float16*)(ws + 0);
    _Float16* XthT = (_Float16*)(ws + 1052672);
    float*    U    = (float*)(ws + 3166208);
    float*    dots = (float*)(ws + 19947520);
    float*    attn = (float*)(ws + 20275200);

    build_atab_kernel<<<512, 256, 0, stream>>>(W, Atab);
    build_xt_kernel<<<1024, 256, 0, stream>>>(X, XthT);
    toeplitz_conv_wmma<<<BS_N * NRT, 128, 0, stream>>>(Atab, XthT, U);
    dim3 dg(T_LEN, BS_N);
    dots_kernel<<<dg, 64, 0, stream>>>(U, X, dots);
    attn_kernel<<<1, 256, 0, stream>>>(dots, C, a, attn);
    out_kernel<<<1, 256, 0, stream>>>(attn, X, out);

    (void)in_sizes; (void)n_in; (void)out_size; (void)ws_size;
}
